// LSTM_12455405159057
// MI455X (gfx1250) — compile-verified, matrix operand order checked by reading, not run on hardware
//
#include <hip/hip_runtime.h>

#define T_STEPS 1024
#define BATCH   512
#define IN_DIM  20
#define H1      128
#define H2      64
#define OUT_DIM 3

typedef __attribute__((ext_vector_type(16))) _Float16 v16h;
typedef __attribute__((ext_vector_type(8)))  _Float16 v8h;
typedef __attribute__((ext_vector_type(8)))  float    v8f;

static __device__ __forceinline__ v8f wmma_f16(v16h a, v16h b, v8f c) {
  // D = A(16x32 f16) * B(32x16 f16) + C(16x16 f32)
  return __builtin_amdgcn_wmma_f32_16x16x32_f16(false, a, false, b, (short)0, c,
                                                false, false);
}

// CDNA5 hardware transcendental tanh (TRANS32). v_nop honors the documented
// 1-instruction result-use gap for trans ops that the compiler cannot see
// through inline asm.
static __device__ __forceinline__ float tanh_hw(float x) {
  float r;
  asm("v_tanh_f32 %0, %1\n\tv_nop" : "=v"(r) : "v"(x));
  return r;
}
static __device__ __forceinline__ float sigmoid_hw(float x) {
  return fmaf(tanh_hw(0.5f * x), 0.5f, 0.5f);   // 1/(1+e^-x) = 0.5+0.5*tanh(x/2)
}

// A-fragment (16x32 f16) per ISA layout: lane L holds row L&15;
// VGPRs 0..3 = K base (+8 if lane>=16), VGPRs 4..7 = K base+16 (+8 if lane>=16).
// p must point at &buf[row][kbase + (lane>=16 ? 8 : 0)], 16B aligned.
static __device__ __forceinline__ v16h load_afrag(const _Float16* p) {
  v8h lo = *(const v8h*)p;
  v8h hv = *(const v8h*)(p + 16);
  v16h a;
#pragma unroll
  for (int e = 0; e < 8; ++e) { a[e] = lo[e]; a[e + 8] = hv[e]; }
  return a;
}

__global__ __launch_bounds__(512, 1)
void lstm_persistent(const float* __restrict__ x,
                     const float* __restrict__ W_ih0, const float* __restrict__ W_hh0,
                     const float* __restrict__ b_ih0, const float* __restrict__ b_hh0,
                     const float* __restrict__ W_ih1, const float* __restrict__ W_hh1,
                     const float* __restrict__ b_ih1, const float* __restrict__ b_hh1,
                     const float* __restrict__ W_d,   const float* __restrict__ b_d,
                     float* __restrict__ out, int mode)
{
  const int tid    = threadIdx.x;
  const int w      = tid >> 5;          // wave id 0..15
  const int lane   = tid & 31;
  const int laneLo = lane & 15;
  const int hi     = (lane >> 4) & 1;
  const int b0     = blockIdx.x * 16;   // batch tile base

  __shared__ __align__(16) _Float16 sX [16][40];   // x_t padded to 32 K (+bank pad)
  __shared__ __align__(16) _Float16 sH0[16][136];  // h0 (128 valid)
  __shared__ __align__(16) _Float16 sH1[16][72];   // h1 (64 valid)
  __shared__ float sG[16][516];                    // gates scratch (max 512 cols)
  __shared__ float sHL[64];                        // mode 1: fp32 h1 of batch row 511
  __shared__ float sHL2[16][64];                   // mode 0: fp32 h1 at last step
  __shared__ float sWd[OUT_DIM][H2];               // dense head weights
  __shared__ float sBd[OUT_DIM];

  // ---------- one-time: weight B-fragments -> registers (f16) ----------
  // B layout (32x16, f16): lane n%16 = column; lanes 0-15 K=0..7,16..23, 16-31 K=8..15,24..31,
  // VGPR v holds K = 2v,2v+1 (+8 for the upper 4 VGPRs, +8 for lanes 16-31). element e: v=e/2, p=e&1.
  v16h w0[2][5];     // 2 col-tiles x 5 K-chunks (chunk 0 = padded x, 1..4 = h0)
#pragma unroll
  for (int tt = 0; tt < 2; ++tt) {
    const int n = 32 * w + 16 * tt + laneLo;   // gate column in [0,512)
#pragma unroll
    for (int kc = 0; kc < 5; ++kc) {
      v16h f;
#pragma unroll
      for (int e = 0; e < 16; ++e) {
        const int v = e >> 1, p = e & 1;
        const int kk = 2 * v + hi * 8 + p + ((v >= 4) ? 8 : 0);
        float val;
        if (kc == 0) val = (kk < IN_DIM) ? W_ih0[n * IN_DIM + kk] : 0.0f;
        else         val = W_hh0[n * H1 + (kc - 1) * 32 + kk];
        f[e] = (_Float16)val;
      }
      w0[tt][kc] = f;
    }
  }
  v16h w1[6];        // 1 col-tile x 6 K-chunks (0..3 = h0, 4..5 = h1)
  {
    const int n = 16 * w + laneLo;             // gate column in [0,256)
#pragma unroll
    for (int kc = 0; kc < 6; ++kc) {
      v16h f;
#pragma unroll
      for (int e = 0; e < 16; ++e) {
        const int v = e >> 1, p = e & 1;
        const int kk = 2 * v + hi * 8 + p + ((v >= 4) ? 8 : 0);
        float val;
        if (kc < 4) val = W_ih1[n * H1 + kc * 32 + kk];
        else        val = W_hh1[n * H2 + (kc - 4) * 32 + kk];
        f[e] = (_Float16)val;
      }
      w1[kc] = f;
    }
  }
  const int   c0a    = 32 * w + laneLo;
  const float bias0a = b_ih0[c0a]      + b_hh0[c0a];
  const float bias0b = b_ih0[c0a + 16] + b_hh0[c0a + 16];
  const int   c1a    = 16 * w + laneLo;
  const float bias1  = b_ih1[c1a] + b_hh1[c1a];

  // ---------- zero LDS state, stage x_0, stage dense head ----------
  for (int i = tid; i < 16 * 40;  i += 512) ((_Float16*)sX )[i] = (_Float16)0.0f;
  for (int i = tid; i < 16 * 136; i += 512) ((_Float16*)sH0)[i] = (_Float16)0.0f;
  for (int i = tid; i < 16 * 72;  i += 512) ((_Float16*)sH1)[i] = (_Float16)0.0f;
  if (tid < OUT_DIM * H2) ((float*)sWd)[tid] = W_d[tid];
  if (tid < OUT_DIM)      sBd[tid] = b_d[tid];
  if (tid < 16 * IN_DIM) {
    const int r = tid / IN_DIM, i = tid % IN_DIM;
    sX[r][i] = (_Float16)x[((size_t)(b0 + r)) * IN_DIM + i];
  }

  float c0s[4] = {0.f, 0.f, 0.f, 0.f};
  float c1s[2] = {0.f, 0.f};
  const bool lastBlk = (blockIdx.x == (BATCH / 16 - 1));

  for (int t = 0; t < T_STEPS; ++t) {
    __syncthreads();   // A: sX(t), sH1(t-1), sHL(t-1) visible; sG free

    // pipelined dense head for step t-1 (mode 1: out is (T,3) from batch row 511)
    if (mode == 1 && lastBlk && t > 0 && tid < OUT_DIM) {
      float s = sBd[tid];
#pragma unroll
      for (int k = 0; k < H2; ++k) s += fmaxf(sHL[k], 0.0f) * sWd[tid][k];
      out[(size_t)(t - 1) * OUT_DIM + tid] = s;
    }

    // -------- layer 0: gates0 = [x_t | h0] @ W0^T + b0 --------
    // -------- plus early layer-1 partial: bias1 + h1(t-1) @ W_hh1^T --------
    v8f acc1p;
    {
      v8f acc0, acc1;
#pragma unroll
      for (int v = 0; v < 8; ++v) { acc0[v] = bias0a; acc1[v] = bias0b; acc1p[v] = bias1; }
      {
        v16h a = load_afrag(&sX[laneLo][hi * 8]);
        acc0 = wmma_f16(a, w0[0][0], acc0);
        acc1 = wmma_f16(a, w0[1][0], acc1);
      }
#pragma unroll
      for (int kc = 1; kc < 5; ++kc) {
        v16h a = load_afrag(&sH0[laneLo][(kc - 1) * 32 + hi * 8]);
        acc0 = wmma_f16(a, w0[0][kc], acc0);
        acc1 = wmma_f16(a, w0[1][kc], acc1);
      }
      // early layer-1 chunks: depend only on h1(t-1), valid since barrier A
#pragma unroll
      for (int kc = 4; kc < 6; ++kc) {
        v16h a = load_afrag(&sH1[laneLo][(kc - 4) * 32 + hi * 8]);
        acc1p = wmma_f16(a, w1[kc], acc1p);
      }
      const int r0 = hi * 8;  // C/D layout: VGPR v -> row v (+8 for lanes 16-31)
#pragma unroll
      for (int v = 0; v < 8; ++v) {
        sG[r0 + v][32 * w + laneLo]      = acc0[v];
        sG[r0 + v][32 * w + 16 + laneLo] = acc1[v];
      }
    }
    __syncthreads();   // B: gates0 ready

    // layer-0 elementwise: 16x128 cells, 4 per thread, c0 state in registers
#pragma unroll
    for (int j = 0; j < 4; ++j) {
      const int idx = tid + 512 * j;
      const int r = idx >> 7, hc = idx & 127;
      const float iv = sigmoid_hw(sG[r][hc]);
      const float fv = sigmoid_hw(sG[r][H1 + hc]);
      const float gv = tanh_hw   (sG[r][2 * H1 + hc]);
      const float ov = sigmoid_hw(sG[r][3 * H1 + hc]);
      c0s[j] = fv * c0s[j] + iv * gv;
      sH0[r][hc] = (_Float16)(ov * tanh_hw(c0s[j]));
    }
    __syncthreads();   // C: h0(t) ready; sG free

    // -------- layer 1 (remaining h0-dependent chunks) --------
    {
      v8f acc = acc1p;
#pragma unroll
      for (int kc = 0; kc < 4; ++kc) {
        v16h a = load_afrag(&sH0[laneLo][kc * 32 + hi * 8]);
        acc = wmma_f16(a, w1[kc], acc);
      }
      const int r0 = hi * 8;
#pragma unroll
      for (int v = 0; v < 8; ++v) sG[r0 + v][16 * w + laneLo] = acc[v];
    }
    __syncthreads();   // D: gates1 ready

    // layer-1 elementwise: 16x64 cells, 2 per thread
#pragma unroll
    for (int j = 0; j < 2; ++j) {
      const int idx = tid + 512 * j;
      const int r = idx >> 6, hc = idx & 63;
      const float iv = sigmoid_hw(sG[r][hc]);
      const float fv = sigmoid_hw(sG[r][H2 + hc]);
      const float gv = tanh_hw   (sG[r][2 * H2 + hc]);
      const float ov = sigmoid_hw(sG[r][3 * H2 + hc]);
      c1s[j] = fv * c1s[j] + iv * gv;
      const float h = ov * tanh_hw(c1s[j]);
      sH1[r][hc] = (_Float16)h;
      if (mode == 1 && r == 15) sHL[hc] = h;                 // batch row 511 (block 31)
      if (mode == 0 && t == T_STEPS - 1) sHL2[r][hc] = h;    // last timestep
    }

    // stage x_{t+1} into LDS; prefetch x_{t+2}
    if (t + 1 < T_STEPS && tid < 16 * IN_DIM) {
      const int r = tid / IN_DIM, i = tid % IN_DIM;
      sX[r][i] = (_Float16)x[((size_t)(t + 1) * BATCH + b0 + r) * IN_DIM + i];
    }
    if (t + 2 < T_STEPS && tid < 16 * IN_DIM) {
      __builtin_prefetch(&x[((size_t)(t + 2) * BATCH + b0 + tid / IN_DIM) * IN_DIM
                            + tid % IN_DIM], 0, 1);
    }
  }

  __syncthreads();
  if (mode == 1) {
    if (lastBlk && tid < OUT_DIM) {                      // final (T-1) row of out
      float s = sBd[tid];
#pragma unroll
      for (int k = 0; k < H2; ++k) s += fmaxf(sHL[k], 0.0f) * sWd[tid][k];
      out[(size_t)(T_STEPS - 1) * OUT_DIM + tid] = s;
    }
  } else {
    if (tid < 16 * OUT_DIM) {                            // out is (B,3), last step
      const int r = tid / OUT_DIM, oc = tid % OUT_DIM;
      float s = sBd[oc];
#pragma unroll
      for (int k = 0; k < H2; ++k) s += fmaxf(sHL2[r][k], 0.0f) * sWd[oc][k];
      out[(size_t)(b0 + r) * OUT_DIM + oc] = s;
    }
  }
}

extern "C" void kernel_launch(void* const* d_in, const int* in_sizes, int n_in,
                              void* d_out, int out_size, void* d_ws, size_t ws_size,
                              hipStream_t stream) {
  const float* x     = (const float*)d_in[0];
  const float* W_ih0 = (const float*)d_in[1];
  const float* W_hh0 = (const float*)d_in[2];
  const float* b_ih0 = (const float*)d_in[3];
  const float* b_hh0 = (const float*)d_in[4];
  const float* W_ih1 = (const float*)d_in[5];
  const float* W_hh1 = (const float*)d_in[6];
  const float* b_ih1 = (const float*)d_in[7];
  const float* b_hh1 = (const float*)d_in[8];
  const float* W_d   = (const float*)d_in[9];
  const float* b_d   = (const float*)d_in[10];
  (void)in_sizes; (void)n_in; (void)d_ws; (void)ws_size;

  // Reference: relu(h2[:, -1, :]) @ W_d.T + b_d with h2 of shape (T,B,H2):
  // out is (T,3)=3072 elements (batch row 511 at every t). Fallback for the
  // (B,3)=1536 "last timestep" interpretation, keyed off out_size (constant
  // per harness -> deterministic).
  const int mode = (out_size == T_STEPS * OUT_DIM) ? 1 : 0;

  hipLaunchKernelGGL(lstm_persistent, dim3(BATCH / 16), dim3(512), 0, stream,
                     x, W_ih0, W_hh0, b_ih0, b_hh0,
                     W_ih1, W_hh1, b_ih1, b_hh1,
                     W_d, b_d, (float*)d_out, mode);
}
